// RelativePositionEmbedding_53060025975544
// MI455X (gfx1250) — matrix operand order repaired and checked by reading, not compiled, not run on hardware
//
#include <hip/hip_runtime.h>
#include <hip/hip_bf16.h>
#include <math.h>

// ---------------------------------------------------------------------------
// MI455X fused Transformer-XL relative-position attention scores.
// Memory-bound on the 512 MiB fp32 output -> single-pass output write (NT hint),
// f16 WMMA (v_wmma_f32_16x16x32_f16) for all GEMMs, S2 strip staged in LDS.
// ---------------------------------------------------------------------------

typedef __attribute__((ext_vector_type(16))) _Float16 v16h;
typedef __attribute__((ext_vector_type(8)))  _Float16 v8h;
typedef __attribute__((ext_vector_type(8)))  float    v8f;

constexpr int Bn  = 2;
constexpr int Tn  = 2048;
constexpr int Hn  = 16;
constexpr int Dn  = 64;
constexpr int DIM = Hn * Dn;          // 1024
constexpr int S2LD = Tn + 4;          // 2052 dwords: +8 rows -> +32 banks, no conflicts
constexpr int S2_BYTES = 17 * S2LD * 4; // 139,536 B dynamic LDS (< 320 KB WGP LDS)

// ---- WMMA operand loaders (layouts per CDNA5 ISA 7.12.2) -------------------
// A (16x32 f16): lane<16 holds row=lane, elems {K0..7, K16..23};
//                lane>=16 holds row=lane-16, elems {K8..15, K24..31}.
__device__ __forceinline__ v16h load_a16(const _Float16* row, int kstep, int lane) {
  const int sel = (lane >> 4) & 1;
  const _Float16* p = row + kstep + sel * 8;
  v8h lo = *(const v8h*)(p);
  v8h hi = *(const v8h*)(p + 16);
  return __builtin_shufflevector(lo, hi, 0,1,2,3,4,5,6,7,8,9,10,11,12,13,14,15);
}
// B (32x16 f16): lane<16 holds col=lane, elems {K0..7, K16..23}; lane>=16 col=lane-16, {K8..15, K24..31}.
__device__ __forceinline__ v16h load_b16(const _Float16* row, int kstep, int lane) {
  const int sel = (lane >> 4) & 1;
  const _Float16* p = row + kstep + sel * 8;
  v8h lo = *(const v8h*)(p);
  v8h hi = *(const v8h*)(p + 16);
  return __builtin_shufflevector(lo, hi, 0,1,2,3,4,5,6,7,8,9,10,11,12,13,14,15);
}
__device__ __forceinline__ v8f wmma_f16(v16h a, v16h b, v8f c) {
  return __builtin_amdgcn_wmma_f32_16x16x32_f16(false, a, false, b, (short)0, c,
                                                false, false);
}

// ---- Prep 1: fold biases, transpose [B,T,H,D] -> [B,H,T,D], fp32 -> f16 ----
__global__ void prep_qk(const float* __restrict__ q, const float* __restrict__ k,
                        const float* __restrict__ rwb, const float* __restrict__ rrb,
                        _Float16* __restrict__ Qw, _Float16* __restrict__ Qr,
                        _Float16* __restrict__ Kc) {
  int idx = blockIdx.x * blockDim.x + threadIdx.x;       // B*T*H*D = 4M
  if (idx >= Bn * Tn * Hn * Dn) return;
  int d = idx & (Dn - 1);
  int h = (idx >> 6) & (Hn - 1);
  int t = (idx >> 10) & (Tn - 1);
  int b = idx >> 21;
  size_t dst = (((size_t)(b * Hn + h) * Tn + t) << 6) + d;
  int hd = (h << 6) + d;
  float qv = q[idx];
  Qw[dst] = (_Float16)(qv + rwb[hd]);
  Qr[dst] = (_Float16)(qv + rrb[hd]);
  Kc[dst] = (_Float16)k[idx];
}

// ---- Prep 2: sinusoidal positional encoding (reversed positions), f16 ------
__global__ void prep_pos(_Float16* __restrict__ P) {
  int idx = blockIdx.x * blockDim.x + threadIdx.x;       // T*DIM = 2M
  if (idx >= Tn * DIM) return;
  int i = idx & (DIM - 1);
  int t = idx >> 10;
  float pos = (float)(Tn - 1 - t);
  int j = (i < DIM / 2) ? i : (i - DIM / 2);
  // inv_freq = 10000^(-2j/dim)
  float inv = expf(-(2.0f * (float)j / (float)DIM) * 9.210340371976184f);
  float v = pos * inv;
  P[idx] = (_Float16)((i < DIM / 2) ? sinf(v) : cosf(v));
}

// ---- Prep 3: transpose w_pos and convert to f16 (B-operand friendly) -------
__global__ void prep_wt(const float* __restrict__ wpos, _Float16* __restrict__ Wt) {
  int idx = blockIdx.x * blockDim.x + threadIdx.x;       // DIM*DIM = 1M
  if (idx >= DIM * DIM) return;
  int kk = idx & (DIM - 1);
  int n  = idx >> 10;
  Wt[idx] = (_Float16)wpos[kk * DIM + n];                // Wt[n][k] = w_pos[k][n]
}

// ---- Prep 4: rel_pos_emb = pos_enc @ w_pos, store as R[H][T][D] f16 --------
__global__ void pos_gemm(const _Float16* __restrict__ P,
                         const _Float16* __restrict__ Wt,
                         _Float16* __restrict__ R) {
  int lane = threadIdx.x & 31;
  int wid  = blockIdx.x * (blockDim.x >> 5) + (threadIdx.x >> 5); // 8192 waves
  int m0 = (wid >> 6) << 4;                              // 128 row tiles
  int n0 = (wid & 63) << 4;                              // 64 col tiles
  int lrow = lane & 15, lsel = (lane >> 4) & 1;
  const _Float16* aRow = P  + (size_t)(m0 + lrow) * DIM;
  const _Float16* bRow = Wt + (size_t)(n0 + lrow) * DIM;
  v8f acc = {};
  for (int kk = 0; kk < DIM; kk += 32)
    acc = wmma_f16(load_a16(aRow, kk, lane), load_b16(bRow, kk, lane), acc);
  int col = n0 + lrow;
  int h = col >> 6, dd = col & 63;
#pragma unroll
  for (int j = 0; j < 8; j++) {
    int t = m0 + j + (lsel << 3);                        // D: vgpr j -> rows j / j+8
    R[(((size_t)h * Tn + t) << 6) + dd] = (_Float16)acc[j];
  }
}

// ---- Main fused kernel: content GEMM + shifted rel GEMM, one output pass ---
// Block = 256 thr = 8 waves; one 16-row strip of one (b,h) per block.
// relative_shift identity: out_rel[i,:] = [S2[i, T-1-i:T], 0, S2[i+1, 0:T-2-i]]
__global__ void relpos_main(const _Float16* __restrict__ Qw,
                            const _Float16* __restrict__ Qr,
                            const _Float16* __restrict__ Kc,
                            const _Float16* __restrict__ R,
                            float* __restrict__ out) {
  extern __shared__ float s2[];                          // [17][S2LD] fp32
  int blk = blockIdx.x;
  int i0  = (blk & 127) << 4;                            // strip base row
  int bh  = blk >> 7;                                    // 0..31
  int h   = bh & (Hn - 1);
  int tid = threadIdx.x, lane = tid & 31, wave = tid >> 5;
  int lrow = lane & 15, lsel = (lane >> 4) & 1;

  const _Float16* qrRow = Qr + (((size_t)bh * Tn + i0 + lrow) << 6);
  int r2 = i0 + 16 + lrow; if (r2 > Tn - 1) r2 = Tn - 1; // clamp; row T never used
  const _Float16* qeRow = Qr + (((size_t)bh * Tn + r2) << 6);
  const _Float16* qwRow = Qw + (((size_t)bh * Tn + i0 + lrow) << 6);
  const _Float16* kBase = Kc + (((size_t)bh * Tn) << 6);
  const _Float16* rBase = R  + (((size_t)h  * Tn) << 6);

  // Qr A-operands for strip rows and the boundary row tile (only M=0 kept).
  v16h ar0 = load_a16(qrRow, 0, lane), ar1 = load_a16(qrRow, 32, lane);
  v16h ae0 = load_a16(qeRow, 0, lane), ae1 = load_a16(qeRow, 32, lane);

  // Phase 1: S2 strip (rows i0..i0+16) x all 2048 cols into LDS.
  for (int ct = wave; ct < Tn / 16; ct += 8) {
    int c0 = ct << 4;
    const _Float16* rRow = rBase + ((size_t)(c0 + lrow) << 6);
    v16h b0 = load_b16(rRow, 0, lane);
    v16h b1 = load_b16(rRow, 32, lane);
    v8f acc  = {}; acc  = wmma_f16(ar0, b0, acc);  acc  = wmma_f16(ar1, b1, acc);
    v8f acce = {}; acce = wmma_f16(ae0, b0, acce); acce = wmma_f16(ae1, b1, acce);
    int col = c0 + lrow;
    int rb  = lsel << 3;
#pragma unroll
    for (int j = 0; j < 8; j++) s2[(rb + j) * S2LD + col] = acc[j];
    if (lsel == 0) s2[16 * S2LD + col] = acce[0];        // boundary row i0+16
  }
  __syncthreads();

  v16h aw0 = load_a16(qwRow, 0, lane), aw1 = load_a16(qwRow, 32, lane);

  // Phase 2: content GEMM + shifted-rel gather from LDS, single output write.
  // Output is write-once streaming (512 MiB, never re-read): use non-temporal
  // stores (TH=NT) so it bypasses L2 residency and keeps the 28 MB operand
  // working set (Kc/R/Qw/Qr) hot in the 192 MB L2 across workgroups.
  for (int ct = wave; ct < Tn / 16; ct += 8) {
    int c0 = ct << 4;
    const _Float16* kRow = kBase + ((size_t)(c0 + lrow) << 6);
    v8f acc = {};
    acc = wmma_f16(aw0, load_b16(kRow, 0, lane), acc);
    acc = wmma_f16(aw1, load_b16(kRow, 32, lane), acc);
    int c  = c0 + lrow;
    int rb = lsel << 3;
#pragma unroll
    for (int j = 0; j < 8; j++) {
      int mloc = rb + j;
      int m    = i0 + mloc;
      int dgap = c - m;
      float rel;
      if (dgap <= 0)      rel = s2[mloc * S2LD + (Tn - 1) + dgap];   // tail of row m
      else if (dgap == 1) rel = 0.0f;                                // pad zero
      else                rel = s2[(mloc + 1) * S2LD + dgap - 2];    // head of row m+1
      __builtin_nontemporal_store(acc[j] + rel,
                                  out + ((size_t)(bh * Tn + m) << 11) + c);
    }
  }
}

// ---------------------------------------------------------------------------
extern "C" void kernel_launch(void* const* d_in, const int* in_sizes, int n_in,
                              void* d_out, int out_size, void* d_ws, size_t ws_size,
                              hipStream_t stream) {
  const float* q    = (const float*)d_in[0];   // [B,T,H,D]
  const float* k    = (const float*)d_in[1];   // [B,T,H,D]
  const float* rwb  = (const float*)d_in[2];   // [H,D]
  const float* rrb  = (const float*)d_in[3];   // [H,D]
  const float* wpos = (const float*)d_in[4];   // [dim,dim]
  float* out = (float*)d_out;                  // [B,H,T,T] fp32

  // Workspace layout (f16), 34 MB total.
  char* w = (char*)d_ws;
  _Float16* Qw = (_Float16*)(w);                               // 8 MB [B,H,T,D]
  _Float16* Qr = (_Float16*)(w + (8u  << 20));                 // 8 MB
  _Float16* Kc = (_Float16*)(w + (16u << 20));                 // 8 MB
  _Float16* P  = (_Float16*)(w + (24u << 20));                 // 4 MB [T,dim]
  _Float16* Wt = (_Float16*)(w + (28u << 20));                 // 2 MB [dim,dim]^T
  _Float16* R  = (_Float16*)(w + (30u << 20));                 // 4 MB [H,T,D]

  prep_qk <<<(Bn * Tn * Hn * Dn) / 256, 256, 0, stream>>>(q, k, rwb, rrb, Qw, Qr, Kc);
  prep_pos<<<(Tn * DIM) / 256,          256, 0, stream>>>(P);
  prep_wt <<<(DIM * DIM) / 256,         256, 0, stream>>>(wpos, Wt);
  pos_gemm<<<(Tn / 16) * (DIM / 16) / 8, 256, 0, stream>>>(P, Wt, R);
  relpos_main<<<Bn * Hn * (Tn / 16), 256, S2_BYTES, stream>>>(Qw, Qr, Kc, R, out);
}
